// DeepNCM_15015205667289
// MI455X (gfx1250) — matrix operand order repaired and checked by reading, not compiled, not run on hardware
//
#include <hip/hip_runtime.h>

// ---------------- problem constants ----------------
#define NUM_CLASSES 1000
#define CPAD        1024      // classes padded to 64*16 WMMA col tiles
#define EMBED_DIM   512
#define NROWS       65536

// ---------------- GEMM tiling ----------------
#define BM   128              // rows per block
#define BN   64               // cols per block
#define BK   32               // k per step (one bf16 WMMA K)
#define ASTR 40               // LDS row stride (bf16 elems), padded vs 32 to kill bank conflicts
#define BSTR 40

typedef __attribute__((ext_vector_type(16))) __bf16 v16bf;
typedef __attribute__((ext_vector_type(8)))  __bf16 v8bf;
typedef __attribute__((ext_vector_type(4)))  __bf16 v4bf;
typedef __attribute__((ext_vector_type(8)))  float  v8f;
typedef __attribute__((ext_vector_type(4)))  int    v4i;

static __device__ __forceinline__ __bf16 f2bf(float x) { return (__bf16)x; }

// ---- gfx1250 async global->LDS path (ASYNCcnt-tracked), guarded fallback ----
#if defined(__HIP_DEVICE_COMPILE__) && __has_builtin(__builtin_amdgcn_global_load_async_to_lds_b128)
#define HAVE_ASYNC_LDS 1
typedef __attribute__((address_space(1))) v4i* as1_v4i_p;  // global (prints as __device__)
typedef __attribute__((address_space(3))) v4i* as3_v4i_p;  // LDS
static __device__ __forceinline__ void async_copy_b128(const void* g, void* l) {
    // AS1 shares address representation with flat for global memory, so an
    // integer round-trip is a valid generic->global cast here.
    __builtin_amdgcn_global_load_async_to_lds_b128(
        (as1_v4i_p)(unsigned long long)g,
        (as3_v4i_p)l,
        0, 0);
}
static __device__ __forceinline__ void async_wait0() {
#if __has_builtin(__builtin_amdgcn_s_wait_asynccnt)
    __builtin_amdgcn_s_wait_asynccnt(0);
#else
    asm volatile("s_wait_asynccnt 0x0" ::: "memory");
#endif
}
#else
#define HAVE_ASYNC_LDS 0
#endif

// ---------------- kernel 1: zero scratch (sums + counts) ----------------
__global__ void k_zero(float* __restrict__ p, int n) {
    int i = blockIdx.x * blockDim.x + threadIdx.x;
    if (i < n) p[i] = 0.0f;
}

// ---------------- kernel 2: segment-sum scatter + ||e||^2 ----------------
// one block (128 threads) per embedding row; fp32 global atomics land in L2
// (sums region is 2MB << 192MB L2).
__global__ void k_scatter(const float* __restrict__ E, const int* __restrict__ y,
                          float* __restrict__ sums, float* __restrict__ counts,
                          float* __restrict__ e_sq) {
    __shared__ float red[128];
    const int n = blockIdx.x;
    const int t = threadIdx.x;
    const int cls = y[n];

    const float4 v = ((const float4*)(E + (size_t)n * EMBED_DIM))[t];
    float* s = sums + (size_t)cls * EMBED_DIM + t * 4;
    atomicAdd(s + 0, v.x);
    atomicAdd(s + 1, v.y);
    atomicAdd(s + 2, v.z);
    atomicAdd(s + 3, v.w);

    red[t] = v.x * v.x + v.y * v.y + v.z * v.z + v.w * v.w;
    __syncthreads();
    for (int off = 64; off > 0; off >>= 1) {
        if (t < off) red[t] += red[t + off];
        __syncthreads();
    }
    if (t == 0) {
        e_sq[n] = red[0];
        atomicAdd(counts + cls, 1.0f);
    }
}

// ---------------- kernel 3: prototype update -> bf16 hi/lo split + ||p||^2 ----
__global__ void k_protos(const float* __restrict__ P, const float* __restrict__ counter,
                         const float* __restrict__ sums, const float* __restrict__ counts,
                         __bf16* __restrict__ Phi, __bf16* __restrict__ Plo,
                         float* __restrict__ p_sq) {
    __shared__ float red[128];
    const int c = blockIdx.x;   // 0..CPAD-1
    const int t = threadIdx.x;  // 128 threads, 4 elems each

    float acc = 0.0f;
    v4bf hi, lo;
    if (c < NUM_CLASSES) {
        const float cnt  = counts[c];
        const float ctr  = counter[c];
        const bool  rep  = cnt > 0.0f;
        const float inv  = 1.0f / fmaxf(cnt, 1.0f);
        const float invc = 1.0f / (ctr + 1.0f);

        const float4 pv = ((const float4*)(P    + (size_t)c * EMBED_DIM))[t];
        const float4 sv = ((const float4*)(sums + (size_t)c * EMBED_DIM))[t];
        float pe[4] = {pv.x, pv.y, pv.z, pv.w};
        float se[4] = {sv.x, sv.y, sv.z, sv.w};
        #pragma unroll
        for (int k = 0; k < 4; ++k) {
            const float np  = se[k] * inv;                 // scatter-mean
            const float upd = (ctr * pe[k] + np) * invc;   // running mean
            const float val = rep ? upd : pe[k];
            acc += val * val;
            const __bf16 h = f2bf(val);
            hi[k] = h;
            lo[k] = f2bf(val - (float)h);                  // bf16x2 split of fp32
        }
    } else {
        #pragma unroll
        for (int k = 0; k < 4; ++k) { hi[k] = f2bf(0.0f); lo[k] = f2bf(0.0f); }
    }
    *(v4bf*)(Phi + (size_t)c * EMBED_DIM + t * 4) = hi;
    *(v4bf*)(Plo + (size_t)c * EMBED_DIM + t * 4) = lo;

    red[t] = acc;
    __syncthreads();
    for (int off = 64; off > 0; off >>= 1) {
        if (t < off) red[t] += red[t + off];
        __syncthreads();
    }
    if (t == 0) p_sq[c] = red[0];
}

// ---------------- kernel 4: WMMA GEMM + distance epilogue ----------------
// out[n][c] = 2*dot(E[n], proto[c]) - ||e_n||^2 - ||p_c||^2
// bf16x3: cross ~= Ehi*Phi + Ehi*Plo + Elo*Phi, accumulated in f32.
__global__ __launch_bounds__(256)
void k_gemm(const float* __restrict__ E,
            const __bf16* __restrict__ Phi, const __bf16* __restrict__ Plo,
            const float* __restrict__ e_sq, const float* __restrict__ p_sq,
            float* __restrict__ out) {
    __shared__ __bf16 Ahi[BM * ASTR];
    __shared__ __bf16 Alo[BM * ASTR];
    __shared__ __bf16 Bhi[BN * BSTR];
    __shared__ __bf16 Blo[BN * BSTR];
    __shared__ float  sEsq[BM];
    __shared__ float  sPsq[BN];

    const int t     = threadIdx.x;
    const int lane  = t & 31;
    const int wave  = t >> 5;     // 0..7
    const int wm    = wave & 3;   // 4 wave rows
    const int wn    = wave >> 2;  // 2 wave cols
    const int rowBase = blockIdx.y * BM;
    const int colBase = blockIdx.x * BN;
    const int m16   = lane & 15;
    const int half  = lane >> 4;

    if (t < BM) sEsq[t] = e_sq[rowBase + t];
    if (t >= 128 && t < 128 + BN) sPsq[t - 128] = p_sq[colBase + (t - 128)];

    v8f acc[2][2] = {};

    for (int kk = 0; kk < EMBED_DIM; kk += BK) {
        __syncthreads();  // previous-iter fragment reads done (also covers sEsq/sPsq init)

        // --- B tile: 64x32 bf16 hi/lo, verbatim copy (never needs VGPRs) ---
        {
            const int r    = t >> 2;
            const int cidx = (t & 3) * 8;
            const __bf16* gh = Phi + (size_t)(colBase + r) * EMBED_DIM + kk + cidx;
            const __bf16* gl = Plo + (size_t)(colBase + r) * EMBED_DIM + kk + cidx;
            __bf16* lh = Bhi + r * BSTR + cidx;
            __bf16* ll = Blo + r * BSTR + cidx;
#if HAVE_ASYNC_LDS
            async_copy_b128(gh, lh);    // ASYNCcnt-tracked, bypasses VGPRs
            async_copy_b128(gl, ll);
#else
            *(v8bf*)lh = *(const v8bf*)gh;
            *(v8bf*)ll = *(const v8bf*)gl;
#endif
        }

        // --- A tile: 128x32 fp32 -> split into bf16 hi/lo in LDS (needs VALU) ---
        #pragma unroll
        for (int p = 0; p < 4; ++p) {
            const int r    = p * 32 + (t >> 3);
            const int cidx = (t & 7) * 4;
            const float4 v = *(const float4*)(E + (size_t)(rowBase + r) * EMBED_DIM + kk + cidx);
            float fe[4] = {v.x, v.y, v.z, v.w};
            v4bf h, l;
            #pragma unroll
            for (int q = 0; q < 4; ++q) {
                const __bf16 hb = f2bf(fe[q]);
                h[q] = hb;
                l[q] = f2bf(fe[q] - (float)hb);
            }
            *(v4bf*)(Ahi + r * ASTR + cidx) = h;
            *(v4bf*)(Alo + r * ASTR + cidx) = l;
        }
#if HAVE_ASYNC_LDS
        async_wait0();   // own async copies landed in LDS
#endif
        __syncthreads(); // everyone's tile writes visible

        // --- fragment loads (ISA 16-bit A 16x32 layout) ---
        // lanes 0-15 (row m): K0-7 in v0-3, K16-23 in v4-7
        // lanes 16-31(row m): K8-15 in v0-3, K24-31 in v4-7
        v16bf afh[2], afl[2], bfh[2], bfl[2];
        #pragma unroll
        for (int i = 0; i < 2; ++i) {
            const int r = wm * 32 + i * 16 + m16;
            const __bf16* pa = Ahi + r * ASTR;
            const __bf16* pl = Alo + r * ASTR;
            v8bf x0 = *(const v8bf*)(pa + half * 8);
            v8bf x1 = *(const v8bf*)(pa + 16 + half * 8);
            v8bf y0 = *(const v8bf*)(pl + half * 8);
            v8bf y1 = *(const v8bf*)(pl + 16 + half * 8);
            #pragma unroll
            for (int q = 0; q < 8; ++q) {
                afh[i][q] = x0[q]; afh[i][8 + q] = x1[q];
                afl[i][q] = y0[q]; afl[i][8 + q] = y1[q];
            }
        }
        // B fragment: lane holds column c = base + m16; K order matches A (K0-7/16-23 | K8-15/24-31)
        #pragma unroll
        for (int j = 0; j < 2; ++j) {
            const int cc = wn * 32 + j * 16 + m16;
            const __bf16* pb = Bhi + cc * BSTR;
            const __bf16* pl = Blo + cc * BSTR;
            v8bf x0 = *(const v8bf*)(pb + half * 8);
            v8bf x1 = *(const v8bf*)(pb + 16 + half * 8);
            v8bf y0 = *(const v8bf*)(pl + half * 8);
            v8bf y1 = *(const v8bf*)(pl + 16 + half * 8);
            #pragma unroll
            for (int q = 0; q < 8; ++q) {
                bfh[j][q] = x0[q]; bfh[j][8 + q] = x1[q];
                bfl[j][q] = y0[q]; bfl[j][8 + q] = y1[q];
            }
        }

        // --- 12 WMMAs per K-step per wave (hi*hi + hi*lo + lo*hi) ---
        #pragma unroll
        for (int i = 0; i < 2; ++i)
            #pragma unroll
            for (int j = 0; j < 2; ++j) {
                acc[i][j] = __builtin_amdgcn_wmma_f32_16x16x32_bf16(
                    false, afh[i], false, bfh[j], (short)0, acc[i][j], false, false);
                acc[i][j] = __builtin_amdgcn_wmma_f32_16x16x32_bf16(
                    false, afh[i], false, bfl[j], (short)0, acc[i][j], false, false);
                acc[i][j] = __builtin_amdgcn_wmma_f32_16x16x32_bf16(
                    false, afl[i], false, bfh[j], (short)0, acc[i][j], false, false);
            }
    }

    // --- epilogue: C/D layout row = e + 8*half, col = lane&15 ---
    // out is a 262MB write-once stream (> L2): use non-temporal stores so L2
    // keeps the E/Phi/Plo read working set that IS reused across column tiles.
    #pragma unroll
    for (int i = 0; i < 2; ++i)
        #pragma unroll
        for (int j = 0; j < 2; ++j) {
            const int cl = wn * 32 + j * 16 + m16;
            const int c  = colBase + cl;
            if (c < NUM_CLASSES) {
                const float psq = sPsq[cl];
                #pragma unroll
                for (int e = 0; e < 8; ++e) {
                    const int rl = wm * 32 + i * 16 + e + 8 * half;
                    const int r  = rowBase + rl;
                    __builtin_nontemporal_store(
                        2.0f * acc[i][j][e] - sEsq[rl] - psq,
                        &out[(size_t)r * NUM_CLASSES + c]);
                }
            }
        }
}

// ---------------- host launcher ----------------
extern "C" void kernel_launch(void* const* d_in, const int* in_sizes, int n_in,
                              void* d_out, int out_size, void* d_ws, size_t ws_size,
                              hipStream_t stream) {
    const float* E       = (const float*)d_in[0];   // [N, D]
    const float* P       = (const float*)d_in[1];   // [C, D]
    const float* counter = (const float*)d_in[2];   // [C]
    const int*   y       = (const int*)d_in[3];     // [N] (jax int64 -> int32 w/o x64)
    float*       out     = (float*)d_out;           // [N, C]

    // workspace carve-up (all 16B-aligned offsets)
    float*  sums   = (float*)d_ws;                               // C*D
    float*  counts = sums + (size_t)NUM_CLASSES * EMBED_DIM;     // CPAD
    float*  e_sq   = counts + CPAD;                              // N
    float*  p_sq   = e_sq + NROWS;                               // CPAD
    __bf16* Phi    = (__bf16*)(p_sq + CPAD);                     // CPAD*D
    __bf16* Plo    = Phi + (size_t)CPAD * EMBED_DIM;             // CPAD*D

    const int nz = NUM_CLASSES * EMBED_DIM + CPAD;
    k_zero<<<dim3((nz + 255) / 256), dim3(256), 0, stream>>>(sums, nz);
    k_scatter<<<dim3(NROWS), dim3(128), 0, stream>>>(E, y, sums, counts, e_sq);
    k_protos<<<dim3(CPAD), dim3(128), 0, stream>>>(P, counter, sums, counts, Phi, Plo, p_sq);
    k_gemm<<<dim3(CPAD / BN, NROWS / BM), dim3(256), 0, stream>>>(E, Phi, Plo, e_sq, p_sq, out);
}
